// DGCNN_cls_19207093748090
// MI455X (gfx1250) — matrix operand order repaired and checked by reading, not compiled, not run on hardware
//
#include <hip/hip_runtime.h>

// ---------------------------------------------------------------------------
// DGCNN-cls forward for MI455X (gfx1250, wave32, WMMA).
// GEMM-heavy ops (edge-convs, 1x1 convs, head linears) use
// v_wmma_f32_16x16x32_f16.  Each 4-wave block computes a 64x64 output tile;
// each wave owns a 32x32 region = 2x2 WMMA tiles -> 4 independent v_wmma per
// 32-deep K chunk with 2x fragment reuse.  The f16 weight tile is staged to
// LDS with GLOBAL_LOAD_ASYNC_TO_LDS_B128 (ASYNCcnt path) when available,
// since it is a raw byte copy needing no VALU conversion; the B tile (f32
// activations / gathered edge features) is converted to f16 via VALU.  The
// edge-conv fuses the neighbor gather, BN + LeakyReLU and the max-over-k
// pooling so the (B,O,N,k) pre-max tensor never touches HBM.  KNN / top-k /
// gathers are simple VALU kernels; their working sets fit in the 192MB L2.
// ---------------------------------------------------------------------------

typedef __attribute__((ext_vector_type(16))) _Float16 v16h;
typedef __attribute__((ext_vector_type(8)))  float    v8f;
typedef __attribute__((ext_vector_type(4)))  int      v4i;

#define NEG_INF (-3.402823466e38f)

#if __has_builtin(__builtin_amdgcn_global_load_async_to_lds_b128) && \
    __has_builtin(__builtin_amdgcn_s_wait_asynccnt)
#define HAVE_ASYNC_LDS 1
#else
#define HAVE_ASYNC_LDS 0
#endif

#define AS_GLOBAL __attribute__((address_space(1)))
#define AS_LDS    __attribute__((address_space(3)))

__device__ __forceinline__ float lrelu(float v) { return v > 0.f ? v : 0.2f * v; }

// async raw-byte stage of a full 64x32-half weight tile into LDS.
// 256 x 16B chunks; 128 threads -> 2 async b128 instructions per wave.
__device__ __forceinline__ void stageA_async(const _Float16* __restrict__ Wh,
                                             _Float16* As, int oBlk, int Ctot,
                                             int c0, int tid) {
#if HAVE_ASYNC_LDS
    for (int e = tid; e < 256; e += 128) {
        int row = e >> 2, part = e & 3;
        const _Float16* g = Wh + (long)(oBlk + row) * Ctot + c0 + part * 8;
        _Float16* l = As + row * 32 + part * 8;
        __builtin_amdgcn_global_load_async_to_lds_b128(
            (AS_GLOBAL v4i*)g, (AS_LDS v4i*)l, 0, 0);
    }
    __builtin_amdgcn_s_wait_asynccnt(0);
#else
    (void)Wh; (void)As; (void)oBlk; (void)Ctot; (void)c0; (void)tid;
#endif
}

// ---------------------------------------------------------------------------
// small utility kernels
// ---------------------------------------------------------------------------
__global__ void cvt_f16_kernel(const float* __restrict__ s, _Float16* __restrict__ d, int n) {
    int i = blockIdx.x * 256 + threadIdx.x;
    if (i < n) d[i] = (_Float16)s[i];
}

// fold eval-mode BN (+ optional linear bias) into per-channel scale/bias
__global__ void prep_bn_kernel(const float* __restrict__ g, const float* __restrict__ b,
                               const float* __restrict__ m, const float* __restrict__ v,
                               const float* __restrict__ linb,
                               float* __restrict__ scale, float* __restrict__ bias, int n) {
    int i = blockIdx.x * 256 + threadIdx.x;
    if (i >= n) return;
    float s = g[i] * rsqrtf(v[i] + 1e-5f);
    scale[i] = s;
    float lb = linb ? linb[i] : 0.f;
    bias[i] = b[i] + (lb - m[i]) * s;
}

// out[b,n] = sum_c x[b,c,n]^2
__global__ void norms_kernel(const float* __restrict__ x, float* __restrict__ out, int C, int N) {
    int n = blockIdx.x * 256 + threadIdx.x;
    int b = blockIdx.y;
    if (n >= N) return;
    const float* xb = x + (long)b * C * N;
    float s = 0.f;
    for (int c = 0; c < C; ++c) { float v = xb[c * N + n]; s += v * v; }
    out[(long)b * N + n] = s;
}

// top-k (largest d = 2*inner - |q|^2 - |r|^2) per query; q==r for self-KNN
__global__ void knn_kernel(const float* __restrict__ q, const float* __restrict__ r,
                           const float* __restrict__ qq, const float* __restrict__ rr,
                           int C, int M, int N, int k, int* __restrict__ idx_out) {
    int m = blockIdx.x * 64 + threadIdx.x;
    int b = blockIdx.y;
    if (m >= M) return;
    const float* qb = q + (long)b * C * M;
    const float* rb = r + (long)b * C * N;
    float qn = qq[(long)b * M + m];
    float bd[32]; int bi[32];
    for (int i = 0; i < k; ++i) { bd[i] = NEG_INF; bi[i] = 0; }
    for (int j = 0; j < N; ++j) {
        float inner = 0.f;
        for (int c = 0; c < C; ++c) inner += qb[c * M + m] * rb[c * N + j];
        float d = 2.f * inner - qn - rr[(long)b * N + j];
        if (d > bd[k - 1]) {
            int p = k - 1;
            while (p > 0 && bd[p - 1] < d) { bd[p] = bd[p - 1]; bi[p] = bi[p - 1]; --p; }
            bd[p] = d; bi[p] = j;
        }
    }
    int* o = idx_out + ((long)b * M + m) * k;
    for (int i = 0; i < k; ++i) o[i] = bi[i];
}

// score[b,n] = w . x[b,:,n] + bias_scalar
__global__ void score_kernel(const float* __restrict__ x, const float* __restrict__ w,
                             const float* __restrict__ bptr, float* __restrict__ out,
                             int C, int N) {
    int n = blockIdx.x * 256 + threadIdx.x;
    int b = blockIdx.y;
    if (n >= N) return;
    const float* xb = x + (long)b * C * N;
    float s = 0.f;
    for (int c = 0; c < C; ++c) s += w[c] * xb[c * N + n];
    out[(long)b * N + n] = s + bptr[0];
}

// iterative block-wide argmax selection of n_out points; writes gate=tanh(val),
// selected indices, and gathers the (3,n_out) static coords into two d_out slices
__global__ void topk_select_kernel(const float* __restrict__ score, float* __restrict__ scratch,
                                   int N, int n_out,
                                   const float* __restrict__ src_xyz, int srcN,
                                   int* __restrict__ idx_out, float* __restrict__ gate_out,
                                   float* __restrict__ ns1, float* __restrict__ ns2) {
    int b = blockIdx.x, tid = threadIdx.x;
    __shared__ float sv[256];
    __shared__ int   si[256];
    float* sc = scratch + (long)b * N;
    for (int i = tid; i < N; i += 256) sc[i] = score[(long)b * N + i];
    __syncthreads();
    for (int t = 0; t < n_out; ++t) {
        float bv = NEG_INF; int bix = 0;
        for (int i = tid; i < N; i += 256) { float v = sc[i]; if (v > bv) { bv = v; bix = i; } }
        sv[tid] = bv; si[tid] = bix;
        __syncthreads();
        for (int s = 128; s > 0; s >>= 1) {
            if (tid < s && sv[tid + s] > sv[tid]) { sv[tid] = sv[tid + s]; si[tid] = si[tid + s]; }
            __syncthreads();
        }
        if (tid == 0) {
            int bid = si[0];
            idx_out[(long)b * n_out + t]  = bid;
            gate_out[(long)b * n_out + t] = tanhf(sv[0]);
            sc[bid] = NEG_INF;
        }
        __syncthreads();
    }
    // gather static coords (3, n_out) into both output slices (n and ns are identical)
    for (int e = tid; e < 3 * n_out; e += 256) {
        int c = e / n_out, t = e % n_out;
        int id = idx_out[(long)b * n_out + t];
        float v = src_xyz[(long)b * 3 * srcN + (long)c * srcN + id];
        ns1[(long)b * 3 * n_out + (long)c * n_out + t] = v;
        ns2[(long)b * 3 * n_out + (long)c * n_out + t] = v;
    }
}

// out[b, c, t] = x[b, c, idx[b,t]] * gate[b,t]   (into a channel-offset slice)
__global__ void gather_scale_kernel(const float* __restrict__ x, const int* __restrict__ idx,
                                    const float* __restrict__ gate, float* __restrict__ out,
                                    int C, int N, int M, int outBatchStride) {
    int e = blockIdx.x * 256 + threadIdx.x;
    int b = blockIdx.y;
    if (e >= C * M) return;
    int c = e / M, t = e % M;
    int id = idx[(long)b * M + t];
    float v = x[(long)b * C * N + (long)c * N + id] * gate[(long)b * M + t];
    out[(long)b * outBatchStride + (long)c * M + t] = v;
}

// out[b, c, m] = max_kk x[b, c, idx[b,m,kk]]
__global__ void agg_max_kernel(const float* __restrict__ x, const int* __restrict__ idx,
                               float* __restrict__ out, int C, int N, int M, int k,
                               int outBatchStride) {
    int e = blockIdx.x * 256 + threadIdx.x;
    int b = blockIdx.y;
    if (e >= C * M) return;
    int c = e / M, m = e % M;
    const int* ib = idx + ((long)b * M + m) * k;
    const float* xb = x + (long)b * C * N + (long)c * N;
    float mx = NEG_INF;
    for (int kk = 0; kk < k; ++kk) mx = fmaxf(mx, xb[ib[kk]]);
    out[(long)b * outBatchStride + (long)c * M + m] = mx;
}

__global__ void resid_lrelu_kernel(const float* __restrict__ a, const float* __restrict__ b,
                                   float* __restrict__ y, int n) {
    int i = blockIdx.x * 256 + threadIdx.x;
    if (i < n) y[i] = lrelu(a[i] + b[i]);
}

// g[b, gOff + c] = max_n x[b, c, n]
__global__ void rowmax_kernel(const float* __restrict__ x, float* __restrict__ g,
                              int C, int N, int gStride, int gOff) {
    int c = blockIdx.x * 256 + threadIdx.x;
    int b = blockIdx.y;
    if (c >= C) return;
    const float* xb = x + (long)b * C * N + (long)c * N;
    float mx = NEG_INF;
    for (int n = 0; n < N; ++n) mx = fmaxf(mx, xb[n]);
    g[(long)b * gStride + gOff + c] = mx;
}

// ---------------------------------------------------------------------------
// WMMA fragment index helpers (ISA 7.12.2 layouts, wave32)
//   A 16x32 f16: M = lane&15; half h -> K = h + 8*(lane>=16) + 8*(h>=8)
//   B 32x16 f16: N = lane&15; half h -> K = h + 8*(lane>=16) + 8*(h>=8)
//   D 16x16 f32: VGPR r -> M = r + 8*(lane>=16), N = lane&15
// ---------------------------------------------------------------------------

// fused edge-conv: gather edge-features + W*feat GEMM (WMMA f16) + BN + lrelu
// + max over k.  Block = 4 waves, tile = 64 O x 64 cols, each wave = 2x2 WMMA
// tiles.  Requires O%64==0, (N*k)%64==0, k|64 (true for every call here).
__global__ void __launch_bounds__(128)
edgeconv_wmma_kernel(const float* __restrict__ x, const int* __restrict__ idx,
                     const _Float16* __restrict__ Wh,
                     const float* __restrict__ scale, const float* __restrict__ bias,
                     float* __restrict__ out, int Cin, int N, int k, int O) {
    const int b = blockIdx.z;
    const int Ctot = 2 * Cin;
    const int oBlk = blockIdx.x * 64;
    const int colBlk = blockIdx.y * 64;
    const int tid = threadIdx.x;
    const int wave = tid >> 5, lane = tid & 31;
    const int oW = (wave >> 1) * 32;   // wave's O sub-block within 64
    const int cW = (wave & 1) * 32;    // wave's col sub-block within 64
    const int laneM = lane & 15, laneHi = lane >> 4;
    const bool asyncA = HAVE_ASYNC_LDS && ((Ctot & 31) == 0);

    __shared__ _Float16 As[64 * 32];   // [o_local][kc]
    __shared__ _Float16 Bs[32 * 64];   // [kc][col_local]
    __shared__ float    Ds[64 * 64];
    __shared__ int mcol[64], ncol[64];

    if (tid < 64) {
        int col = colBlk + tid;
        int n = col / k, kk = col % k;
        ncol[tid] = n;
        mcol[tid] = idx[((long)b * N + n) * k + kk];
    }
    __syncthreads();

    const float* xb = x + (long)b * Cin * N;
    v8f acc00, acc01, acc10, acc11;
#pragma unroll
    for (int i = 0; i < 8; ++i) { acc00[i] = 0.f; acc01[i] = 0.f; acc10[i] = 0.f; acc11[i] = 0.f; }

    for (int c0 = 0; c0 < Ctot; c0 += 32) {
        // prefetch next K-chunk of the streaming weight rows (global_prefetch_b8)
        if (c0 + 32 < Ctot)
            __builtin_prefetch(Wh + (long)(oBlk + (tid >> 1)) * Ctot + c0 + 32, 0, 3);
        // stage A (weights, raw f16 bytes) tile: 64x32 halfs
        if (asyncA) {
            stageA_async(Wh, As, oBlk, Ctot, c0, tid);
        } else {
            for (int e = tid; e < 64 * 32; e += 128) {
                int ol = e >> 5, kc = e & 31;
                int c = c0 + kc;
                As[e] = (c < Ctot) ? Wh[(long)(oBlk + ol) * Ctot + c] : (_Float16)0.f;
            }
        }
        // stage B (gathered edge features, f32 -> f16) tile: 32x64 halfs
        for (int e = tid; e < 32 * 64; e += 128) {
            int kc = e >> 6, j = e & 63;
            int c = c0 + kc;
            float v = 0.f;
            if (c < Cin)       v = xb[(long)c * N + mcol[j]] - xb[(long)c * N + ncol[j]];
            else if (c < Ctot) v = xb[(long)(c - Cin) * N + ncol[j]];
            Bs[kc * 64 + j] = (_Float16)v;
        }
        __syncthreads();
        v16h a0, a1, bf0, bf1;
#pragma unroll
        for (int h = 0; h < 16; ++h) {
            int ka = h + (laneHi ? 8 : 0) + (h >= 8 ? 8 : 0);
            a0[h]  = As[(oW + laneM) * 32 + ka];
            a1[h]  = As[(oW + 16 + laneM) * 32 + ka];
            int kb = h + (laneHi ? 8 : 0) + (h >= 8 ? 8 : 0);
            bf0[h] = Bs[kb * 64 + cW + laneM];
            bf1[h] = Bs[kb * 64 + cW + 16 + laneM];
        }
        acc00 = __builtin_amdgcn_wmma_f32_16x16x32_f16(false, a0, false, bf0, (short)0, acc00, false, false);
        acc01 = __builtin_amdgcn_wmma_f32_16x16x32_f16(false, a0, false, bf1, (short)0, acc01, false, false);
        acc10 = __builtin_amdgcn_wmma_f32_16x16x32_f16(false, a1, false, bf0, (short)0, acc10, false, false);
        acc11 = __builtin_amdgcn_wmma_f32_16x16x32_f16(false, a1, false, bf1, (short)0, acc11, false, false);
        __syncthreads();
    }

    // epilogue: BN + lrelu into LDS
#pragma unroll
    for (int r = 0; r < 8; ++r) {
        int mr = r + laneHi * 8;
        {
            int ol = oW + mr; int o = oBlk + ol;
            Ds[ol * 64 + cW + laneM]      = lrelu(acc00[r] * scale[o] + bias[o]);
            Ds[ol * 64 + cW + 16 + laneM] = lrelu(acc01[r] * scale[o] + bias[o]);
        }
        {
            int ol = oW + 16 + mr; int o = oBlk + ol;
            Ds[ol * 64 + cW + laneM]      = lrelu(acc10[r] * scale[o] + bias[o]);
            Ds[ol * 64 + cW + 16 + laneM] = lrelu(acc11[r] * scale[o] + bias[o]);
        }
    }
    __syncthreads();

    // max over the k neighbors of each point owned by this block
    int nPer = 64 / k;
    int nBase = colBlk / k;
    for (int e = tid; e < 64 * nPer; e += 128) {
        int ol = e / nPer, nl = e % nPer;
        float mx = NEG_INF;
        for (int kk = 0; kk < k; ++kk) mx = fmaxf(mx, Ds[ol * 64 + nl * k + kk]);
        out[((long)b * O + oBlk + ol) * N + nBase + nl] = mx;
    }
}

// generic strided GEMM: out(o,j) = act( scale[o] * sum_c W[o,c] X(c,j) + bias[o] )
// serves 1x1 convs over (B,C,N) and the head linears (cols = batch).  Same
// 64x64 / 2x2-tiles-per-wave structure, with O / cols guards for the head.
__global__ void __launch_bounds__(128)
gemm_wmma_kernel(const float* __restrict__ X, const _Float16* __restrict__ Wh,
                 const float* __restrict__ scale, const float* __restrict__ bias,
                 float* __restrict__ out, int Ctot, int M, int O,
                 long xsC, long xsJ, long xsB,
                 long osO, long osJ, long osB, int act) {
    const int b = blockIdx.z;
    const int oBlk = blockIdx.x * 64;
    const int colBlk = blockIdx.y * 64;
    const int tid = threadIdx.x;
    const int wave = tid >> 5, lane = tid & 31;
    const int oW = (wave >> 1) * 32;
    const int cW = (wave & 1) * 32;
    const int laneM = lane & 15, laneHi = lane >> 4;
    const bool asyncA = HAVE_ASYNC_LDS && ((Ctot & 31) == 0) && (oBlk + 64 <= O);

    __shared__ _Float16 As[64 * 32];
    __shared__ _Float16 Bs[32 * 64];

    const float* xb = X + (long)b * xsB;
    v8f acc00, acc01, acc10, acc11;
#pragma unroll
    for (int i = 0; i < 8; ++i) { acc00[i] = 0.f; acc01[i] = 0.f; acc10[i] = 0.f; acc11[i] = 0.f; }

    for (int c0 = 0; c0 < Ctot; c0 += 32) {
        if (c0 + 32 < Ctot)
            __builtin_prefetch(Wh + (long)(oBlk + (tid >> 1)) * Ctot + c0 + 32, 0, 3);
        if (asyncA) {
            stageA_async(Wh, As, oBlk, Ctot, c0, tid);
        } else {
            for (int e = tid; e < 64 * 32; e += 128) {
                int ol = e >> 5, kc = e & 31;
                int c = c0 + kc, orow = oBlk + ol;
                As[e] = (orow < O && c < Ctot) ? Wh[(long)orow * Ctot + c] : (_Float16)0.f;
            }
        }
        for (int e = tid; e < 32 * 64; e += 128) {
            int kc = e >> 6, jl = e & 63;
            int c = c0 + kc, j = colBlk + jl;
            float v = (c < Ctot && j < M) ? xb[c * xsC + j * xsJ] : 0.f;
            Bs[kc * 64 + jl] = (_Float16)v;
        }
        __syncthreads();
        v16h a0, a1, bf0, bf1;
#pragma unroll
        for (int h = 0; h < 16; ++h) {
            int ka = h + (laneHi ? 8 : 0) + (h >= 8 ? 8 : 0);
            a0[h]  = As[(oW + laneM) * 32 + ka];
            a1[h]  = As[(oW + 16 + laneM) * 32 + ka];
            int kb = h + (laneHi ? 8 : 0) + (h >= 8 ? 8 : 0);
            bf0[h] = Bs[kb * 64 + cW + laneM];
            bf1[h] = Bs[kb * 64 + cW + 16 + laneM];
        }
        acc00 = __builtin_amdgcn_wmma_f32_16x16x32_f16(false, a0, false, bf0, (short)0, acc00, false, false);
        acc01 = __builtin_amdgcn_wmma_f32_16x16x32_f16(false, a0, false, bf1, (short)0, acc01, false, false);
        acc10 = __builtin_amdgcn_wmma_f32_16x16x32_f16(false, a1, false, bf0, (short)0, acc10, false, false);
        acc11 = __builtin_amdgcn_wmma_f32_16x16x32_f16(false, a1, false, bf1, (short)0, acc11, false, false);
        __syncthreads();
    }

    auto st = [&](int o, int j, float v) {
        if (o < O && j < M) {
            if (scale) v *= scale[o];
            if (bias)  v += bias[o];
            if (act)   v = lrelu(v);
            out[(long)b * osB + o * osO + j * osJ] = v;
        }
    };
#pragma unroll
    for (int r = 0; r < 8; ++r) {
        int mr = r + laneHi * 8;
        int o0 = oBlk + oW + mr, o1 = o0 + 16;
        int j0 = colBlk + cW + laneM, j1 = j0 + 16;
        st(o0, j0, acc00[r]);
        st(o0, j1, acc01[r]);
        st(o1, j0, acc10[r]);
        st(o1, j1, acc11[r]);
    }
}

// ---------------------------------------------------------------------------
// host orchestration
// ---------------------------------------------------------------------------
extern "C" void kernel_launch(void* const* d_in, const int* in_sizes, int n_in,
                              void* d_out, int out_size, void* d_ws, size_t ws_size,
                              hipStream_t stream) {
    (void)in_sizes; (void)n_in; (void)out_size; (void)ws_size;
    const int B = 8;

    // ---- workspace bump allocator ----
    size_t off = 0;
    auto alloc = [&](size_t bytes) -> void* {
        void* p = (char*)d_ws + off;
        off += (bytes + 255) & ~(size_t)255;
        return p;
    };
    auto F = [&](const void* p) { return (const float*)p; };

    // ---- convert weights to f16, fold BN params ----
    static const int wIdx[17] = {1, 6, 11, 16, 21, 26, 31, 36, 41, 46, 51, 56, 61, 66, 77, 82, 88};
    static const int wO[17]   = {64, 64, 128, 128, 256, 256, 512, 512, 512, 512, 512, 128, 256, 512, 512, 256, 40};
    static const int wC[17]   = {6, 128, 256, 256, 512, 512, 1024, 1024, 64, 128, 256, 128, 256, 512, 2048, 512, 256};
    _Float16* Wh[17];
    for (int i = 0; i < 17; ++i) {
        int n = wO[i] * wC[i];
        Wh[i] = (_Float16*)alloc((size_t)n * sizeof(_Float16));
        cvt_f16_kernel<<<dim3((n + 255) / 256), 256, 0, stream>>>(F(d_in[wIdx[i]]), Wh[i], n);
    }
    float *Sc[16], *Bi[16];
    for (int i = 0; i < 14; ++i) {  // 14 conv BNs follow each weight
        int O = wO[i];
        Sc[i] = (float*)alloc(O * 4); Bi[i] = (float*)alloc(O * 4);
        prep_bn_kernel<<<dim3((O + 255) / 256), 256, 0, stream>>>(
            F(d_in[wIdx[i] + 1]), F(d_in[wIdx[i] + 2]), F(d_in[wIdx[i] + 3]), F(d_in[wIdx[i] + 4]),
            nullptr, Sc[i], Bi[i], O);
    }
    Sc[14] = (float*)alloc(512 * 4); Bi[14] = (float*)alloc(512 * 4);   // bn9 (lin1)
    prep_bn_kernel<<<dim3(2), 256, 0, stream>>>(F(d_in[78]), F(d_in[79]), F(d_in[80]), F(d_in[81]),
                                                nullptr, Sc[14], Bi[14], 512);
    Sc[15] = (float*)alloc(256 * 4); Bi[15] = (float*)alloc(256 * 4);   // bn10 (lin2, + lin2_b)
    prep_bn_kernel<<<dim3(1), 256, 0, stream>>>(F(d_in[84]), F(d_in[85]), F(d_in[86]), F(d_in[87]),
                                                F(d_in[83]), Sc[15], Bi[15], 256);

    // ---- activation buffers ----
    float* normQ = (float*)alloc((size_t)B * 2048 * 4);
    float* normR = (float*)alloc((size_t)B * 2048 * 4);
    int*   idxb  = (int*)alloc((size_t)B * 2048 * 32 * 4);
    float* scr   = (float*)alloc((size_t)B * 2048 * 4);
    int*   selIdx= (int*)alloc((size_t)B * 256 * 4);
    float* gate  = (float*)alloc((size_t)B * 256 * 4);
    float* x1    = (float*)alloc((size_t)B * 64 * 2048 * 4);
    float* x2    = (float*)alloc((size_t)B * 64 * 2048 * 4);
    float* xt1   = (float*)alloc((size_t)B * 512 * 2048 * 4);
    float* cat1  = (float*)alloc((size_t)B * 128 * 256 * 4);
    float* xp1   = (float*)alloc((size_t)B * 128 * 256 * 4);
    float* x3    = (float*)alloc((size_t)B * 128 * 256 * 4);
    float* t1    = (float*)alloc((size_t)B * 128 * 256 * 4);
    float* x4    = (float*)alloc((size_t)B * 128 * 256 * 4);
    float* xt2   = (float*)alloc((size_t)B * 512 * 256 * 4);
    float* cat2  = (float*)alloc((size_t)B * 256 * 64 * 4);
    float* xp2   = (float*)alloc((size_t)B * 256 * 64 * 4);
    float* x5    = (float*)alloc((size_t)B * 256 * 64 * 4);
    float* t2    = (float*)alloc((size_t)B * 256 * 64 * 4);
    float* x6    = (float*)alloc((size_t)B * 256 * 64 * 4);
    float* xt3   = (float*)alloc((size_t)B * 512 * 64 * 4);
    float* cat3  = (float*)alloc((size_t)B * 512 * 16 * 4);
    float* xp3   = (float*)alloc((size_t)B * 512 * 16 * 4);
    float* x7    = (float*)alloc((size_t)B * 512 * 16 * 4);
    float* t3    = (float*)alloc((size_t)B * 512 * 16 * 4);
    float* xt4   = (float*)alloc((size_t)B * 512 * 16 * 4);
    float* gbuf  = (float*)alloc((size_t)B * 2048 * 4);
    float* h1    = (float*)alloc((size_t)B * 512 * 4);
    float* h2    = (float*)alloc((size_t)B * 256 * 4);

    // ---- output slices (return-order flattening) ----
    float* outp   = (float*)d_out;
    float* o_log  = outp;              // (8,40)
    float* o_ret1 = outp + 320;        // (8,2048)
    float* o_ret2 = o_ret1 + 16384;    // (8,256)
    float* o_ret3 = o_ret2 + 2048;     // (8,64)
    float* o_n1   = o_ret3 + 512;      // (8,3,256)
    float* o_n2   = o_n1 + 6144;       // (8,3,64)
    float* o_n3   = o_n2 + 1536;       // (8,3,16)
    float* o_n1s  = o_n3 + 384;
    float* o_n2s  = o_n1s + 6144;
    float* o_n3s  = o_n2s + 1536;

    // ---- launch helpers ----
    auto NORMS = [&](const float* x, float* o, int C, int N) {
        norms_kernel<<<dim3((N + 255) / 256, B), 256, 0, stream>>>(x, o, C, N);
    };
    auto KNN = [&](const float* q, const float* r, const float* qq, const float* rr,
                   int C, int M, int N, int k) {
        knn_kernel<<<dim3((M + 63) / 64, B), 64, 0, stream>>>(q, r, qq, rr, C, M, N, k, idxb);
    };
    auto EDGE = [&](const float* x, const _Float16* W, const float* s, const float* bi,
                    float* o, int Cin, int N, int k, int O) {
        edgeconv_wmma_kernel<<<dim3(O / 64, (N * k) / 64, B), 128, 0, stream>>>(
            x, idxb, W, s, bi, o, Cin, N, k, O);
    };
    auto CONV1D = [&](const float* X, const _Float16* W, const float* s, const float* bi,
                      float* o, int C, int N, int O) {
        gemm_wmma_kernel<<<dim3((O + 63) / 64, (N + 63) / 64, B), 128, 0, stream>>>(
            X, W, s, bi, o, C, N, O, (long)N, 1L, (long)C * N, (long)N, 1L, (long)O * N, 1);
    };
    auto LINEAR = [&](const float* X, const _Float16* W, const float* s, const float* bi,
                      float* o, int C, int O, int act) {
        gemm_wmma_kernel<<<dim3((O + 63) / 64, 1, 1), 128, 0, stream>>>(
            X, W, s, bi, o, C, B, O, 1L, (long)C, 0L, 1L, (long)O, 0L, act);
    };

    const float* xin = F(d_in[0]);

    // ---------------- stage 1 (2048 pts, k=32) ----------------
    NORMS(xin, normR, 3, 2048);
    KNN(xin, xin, normR, normR, 3, 2048, 2048, 32);
    EDGE(xin, Wh[0], Sc[0], Bi[0], x1, 3, 2048, 32, 64);          // conv1
    NORMS(x1, normR, 64, 2048);
    KNN(x1, x1, normR, normR, 64, 2048, 2048, 32);
    EDGE(x1, Wh[1], Sc[1], Bi[1], x2, 64, 2048, 32, 64);          // conv2
    CONV1D(x2, Wh[8], Sc[8], Bi[8], xt1, 64, 2048, 512);          // conv2_m
    score_kernel<<<dim3(8, B), 256, 0, stream>>>(x2, F(d_in[71]), F(d_in[72]), o_ret1, 64, 2048);
    topk_select_kernel<<<dim3(B), 256, 0, stream>>>(o_ret1, scr, 2048, 256, xin, 2048,
                                                    selIdx, gate, o_n1, o_n1s);
    gather_scale_kernel<<<dim3((64 * 256 + 255) / 256, B), 256, 0, stream>>>(
        x2, selIdx, gate, cat1, 64, 2048, 256, 2 * 64 * 256);
    NORMS(o_n1s, normQ, 3, 256);
    NORMS(xin, normR, 3, 2048);
    KNN(o_n1s, xin, normQ, normR, 3, 256, 2048, 32);
    agg_max_kernel<<<dim3((64 * 256 + 255) / 256, B), 256, 0, stream>>>(
        x2, idxb, cat1 + 64 * 256, 64, 2048, 256, 32, 2 * 64 * 256);
    CONV1D(cat1, Wh[11], Sc[11], Bi[11], xp1, 128, 256, 128);     // conv2_p

    // ---------------- stage 2 (256 pts, k=16) ----------------
    NORMS(xp1, normR, 128, 256);
    KNN(xp1, xp1, normR, normR, 128, 256, 256, 16);
    EDGE(xp1, Wh[2], Sc[2], Bi[2], x3, 128, 256, 16, 128);        // conv3
    NORMS(x3, normR, 128, 256);
    KNN(x3, x3, normR, normR, 128, 256, 256, 16);
    EDGE(x3, Wh[3], Sc[3], Bi[3], t1, 128, 256, 16, 128);         // conv4
    resid_lrelu_kernel<<<dim3((B * 128 * 256 + 255) / 256), 256, 0, stream>>>(t1, xp1, x4, B * 128 * 256);
    CONV1D(x4, Wh[9], Sc[9], Bi[9], xt2, 128, 256, 512);          // conv4_m
    score_kernel<<<dim3(1, B), 256, 0, stream>>>(x4, F(d_in[73]), F(d_in[74]), o_ret2, 128, 256);
    topk_select_kernel<<<dim3(B), 256, 0, stream>>>(o_ret2, scr, 256, 64, o_n1s, 256,
                                                    selIdx, gate, o_n2, o_n2s);
    gather_scale_kernel<<<dim3((128 * 64 + 255) / 256, B), 256, 0, stream>>>(
        x4, selIdx, gate, cat2, 128, 256, 64, 2 * 128 * 64);
    NORMS(o_n2s, normQ, 3, 64);
    NORMS(o_n1s, normR, 3, 256);
    KNN(o_n2s, o_n1s, normQ, normR, 3, 64, 256, 16);
    agg_max_kernel<<<dim3((128 * 64 + 255) / 256, B), 256, 0, stream>>>(
        x4, idxb, cat2 + 128 * 64, 128, 256, 64, 16, 2 * 128 * 64);
    CONV1D(cat2, Wh[12], Sc[12], Bi[12], xp2, 256, 64, 256);      // conv4_p

    // ---------------- stage 3 (64 pts, k=8; aggregate uses k=32) ----------------
    NORMS(xp2, normR, 256, 64);
    KNN(xp2, xp2, normR, normR, 256, 64, 64, 8);
    EDGE(xp2, Wh[4], Sc[4], Bi[4], x5, 256, 64, 8, 256);          // conv5
    NORMS(x5, normR, 256, 64);
    KNN(x5, x5, normR, normR, 256, 64, 64, 8);
    EDGE(x5, Wh[5], Sc[5], Bi[5], t2, 256, 64, 8, 256);           // conv6
    resid_lrelu_kernel<<<dim3((B * 256 * 64 + 255) / 256), 256, 0, stream>>>(t2, xp2, x6, B * 256 * 64);
    CONV1D(x6, Wh[10], Sc[10], Bi[10], xt3, 256, 64, 512);        // conv6_m
    score_kernel<<<dim3(1, B), 256, 0, stream>>>(x6, F(d_in[75]), F(d_in[76]), o_ret3, 256, 64);
    topk_select_kernel<<<dim3(B), 256, 0, stream>>>(o_ret3, scr, 64, 16, o_n2s, 64,
                                                    selIdx, gate, o_n3, o_n3s);
    gather_scale_kernel<<<dim3((256 * 16 + 255) / 256, B), 256, 0, stream>>>(
        x6, selIdx, gate, cat3, 256, 64, 16, 2 * 256 * 16);
    NORMS(o_n3s, normQ, 3, 16);
    NORMS(o_n2s, normR, 3, 64);
    KNN(o_n3s, o_n2s, normQ, normR, 3, 16, 64, 32);
    agg_max_kernel<<<dim3((256 * 16 + 255) / 256, B), 256, 0, stream>>>(
        x6, idxb, cat3 + 256 * 16, 256, 64, 16, 32, 2 * 256 * 16);
    CONV1D(cat3, Wh[13], Sc[13], Bi[13], xp3, 512, 16, 512);      // conv6_p

    // ---------------- stage 4 (16 pts, k=4) ----------------
    NORMS(xp3, normR, 512, 16);
    KNN(xp3, xp3, normR, normR, 512, 16, 16, 4);
    EDGE(xp3, Wh[6], Sc[6], Bi[6], x7, 512, 16, 4, 512);          // conv7
    NORMS(x7, normR, 512, 16);
    KNN(x7, x7, normR, normR, 512, 16, 16, 4);
    EDGE(x7, Wh[7], Sc[7], Bi[7], t3, 512, 16, 4, 512);           // conv8
    resid_lrelu_kernel<<<dim3((B * 512 * 16 + 255) / 256), 256, 0, stream>>>(t3, xp3, xt4, B * 512 * 16);

    // ---------------- head ----------------
    rowmax_kernel<<<dim3(2, B), 256, 0, stream>>>(xt1, gbuf, 512, 2048, 2048, 0);
    rowmax_kernel<<<dim3(2, B), 256, 0, stream>>>(xt2, gbuf, 512, 256, 2048, 512);
    rowmax_kernel<<<dim3(2, B), 256, 0, stream>>>(xt3, gbuf, 512, 64, 2048, 1024);
    rowmax_kernel<<<dim3(2, B), 256, 0, stream>>>(xt4, gbuf, 512, 16, 2048, 1536);
    LINEAR(gbuf, Wh[14], Sc[14], Bi[14], h1, 2048, 512, 1);       // lin1 + bn9 + lrelu
    LINEAR(h1,   Wh[15], Sc[15], Bi[15], h2, 512, 256, 1);        // lin2 + bn10 + lrelu
    LINEAR(h2,   Wh[16], nullptr, F(d_in[89]), o_log, 256, 40, 0);// lin3 + bias
}
